// TransofmerEncoderBlock_29222957482015
// MI455X (gfx1250) — matrix operand order repaired and checked by reading, not compiled, not run on hardware
//
#include <hip/hip_runtime.h>
#include <hip/hip_bf16.h>
#include <math.h>

// ---------------------------------------------------------------------------
// Problem constants (from reference): B=8, S=1024, D=768, H=12, HD=64, HID=3072
// ---------------------------------------------------------------------------
#define BB   8
#define SS   1024
#define DD   768
#define HH   12
#define HDD  64
#define HID  3072
#define ROWS (BB * SS)          // 8192

typedef __attribute__((ext_vector_type(16))) _Float16 v16h;
typedef __attribute__((ext_vector_type(8)))  _Float16 v8h;
typedef __attribute__((ext_vector_type(8)))  float    v8f;

union HFrag { v16h v; v8h h[2]; _Float16 e[16]; };

__device__ inline v8f wmma16(v16h a, v16h b, v8f c) {
  // D = A(16x32 f16) * B(32x16 f16) + C(16x16 f32)
  return __builtin_amdgcn_wmma_f32_16x16x32_f16(
      /*neg_a=*/false, a, /*neg_b=*/false, b,
      /*c_mod=*/(short)0, c, /*reuse_a=*/false, /*reuse_b=*/false);
}

__device__ inline float gelu_exact(float x) {
  return 0.5f * x * (1.0f + erff(x * 0.70710678118654752f));
}

// --- CDNA5 async copy: global -> LDS, tracked by ASYNCcnt -------------------
// LDS address from the vdst VGPR, global address from vaddr; offset: adds to both
__device__ inline void async_ld_b128(unsigned ldsaddr, const void* gaddr) {
  asm volatile("global_load_async_to_lds_b128 %0, %1, off"
               :: "v"(ldsaddr), "v"(gaddr) : "memory");
}
__device__ inline void async_ld_b128_o16(unsigned ldsaddr, const void* gaddr) {
  asm volatile("global_load_async_to_lds_b128 %0, %1, off offset:16"
               :: "v"(ldsaddr), "v"(gaddr) : "memory");
}
__device__ inline void wait_async0() {
  asm volatile("s_wait_asynccnt 0x0" ::: "memory");
}
__device__ inline unsigned lds_off(const void* p) {
  return (unsigned)(uintptr_t)p;   // low 32 bits of generic LDS address
}

// ---------------------------------------------------------------------------
// f32 -> f16 cast (plain)
// ---------------------------------------------------------------------------
__global__ void cast_f32_to_f16(const float* __restrict__ in,
                                _Float16* __restrict__ out, long n) {
  long i = (long)blockIdx.x * blockDim.x + threadIdx.x;
  long stride = (long)gridDim.x * blockDim.x;
  for (; i < n; i += stride) out[i] = (_Float16)in[i];
}

// ---------------------------------------------------------------------------
// f32 -> f16 cast + transpose:  in[K][N] (f32)  ->  out[N][K] (f16)
// 32x32 tiles via LDS, 256 threads (32x8).
// ---------------------------------------------------------------------------
__global__ void cast_transpose_f32_to_f16(const float* __restrict__ in,
                                          _Float16* __restrict__ out,
                                          int K, int N) {
  __shared__ float tile[32][33];
  const int nt = blockIdx.x * 32;
  const int kt = blockIdx.y * 32;
  const int tx = threadIdx.x & 31;
  const int ty = threadIdx.x >> 5;     // 0..7
#pragma unroll
  for (int i = 0; i < 32; i += 8)
    tile[ty + i][tx] = in[(size_t)(kt + ty + i) * N + nt + tx];
  __syncthreads();
#pragma unroll
  for (int i = 0; i < 32; i += 8)
    out[(size_t)(nt + ty + i) * K + kt + tx] = (_Float16)tile[tx][ty + i];
}

// ---------------------------------------------------------------------------
// Generic WMMA GEMM: out = epilogue(A[M,K](f16) @ WT[N,K](f16, pre-transposed))
//   block tile 128x128, 8 waves (4 in M x 2 in N), wave tile 32x64 (2x4 WMMA)
//   => 8 WMMAs per 12 ds_load_b128 per wave per K-step.
//   double-buffered LDS; BOTH tiles staged with global_load_async_to_lds_b128
//   (no ds_store in the loop at all).
//   epilogue: + bias, optional exact GELU, optional residual add,
//             f32 / f16 / f16-transposed([row/1024][N][1024]) stores.
// M % 128 == 0, N % 128 == 0, K % 32 == 0 (true for all instances here)
// ---------------------------------------------------------------------------
__global__ void gemm_wmma_f16(const _Float16* __restrict__ A,
                              const _Float16* __restrict__ WT,
                              const float* __restrict__ bias,
                              float* __restrict__ outF32,
                              _Float16* __restrict__ outF16,
                              _Float16* __restrict__ outF16T,
                              const float* __restrict__ residual,
                              int M, int N, int K, int doGelu) {
  const int n0 = blockIdx.x * 128;
  const int m0 = blockIdx.y * 128;
  const int t = threadIdx.x;
  const int wave = t >> 5;
  const int lane = t & 31;
  const int lam  = lane & 15;
  const int hi   = lane >> 4;
  const int wm   = wave & 3;   // 0..3 -> M group (32 rows)
  const int wn   = wave >> 2;  // 0..1 -> N group (64 cols)

  __shared__ __align__(16) _Float16 As[2][128][32];   // 16 KB
  __shared__ __align__(16) _Float16 Bt[2][128][32];   // 16 KB, [ncol][k]

  // per-thread staging coordinates (each thread: 2 b128 for A, 2 for B)
  const int ar = t >> 1;            // A row 0..127
  const int ac = (t & 1) * 16;      // A col 0 or 16
  const int bc = t >> 1;            // B ncol 0..127
  const int bs = (t & 1) * 16;      // B k 0 or 16

  const _Float16* srcA0 = A + (size_t)(m0 + ar) * K + ac;
  const _Float16* srcB0 = WT + (size_t)(n0 + bc) * K + bs;

  v8f zero = {};
  v8f acc[2][4];
#pragma unroll
  for (int i = 0; i < 2; ++i)
#pragma unroll
    for (int j = 0; j < 4; ++j) acc[i][j] = zero;

  // ---- stage first tile (all async) ----
  {
    const unsigned la = lds_off(&As[0][ar][ac]);
    async_ld_b128(la, srcA0);
    async_ld_b128_o16(la, srcA0);
    const unsigned lb = lds_off(&Bt[0][bc][bs]);
    async_ld_b128(lb, srcB0);
    async_ld_b128_o16(lb, srcB0);
  }

  int buf = 0;
  for (int k0 = 0; k0 < K; k0 += 32, buf ^= 1) {
    wait_async0();
    __syncthreads();

    // ---- stage next tile into the other buffer (overlaps with WMMA) ----
    if (k0 + 32 < K) {
      const int kn = k0 + 32;
      const unsigned la = lds_off(&As[buf ^ 1][ar][ac]);
      async_ld_b128(la, srcA0 + kn);
      async_ld_b128_o16(la, srcA0 + kn);
      const unsigned lb = lds_off(&Bt[buf ^ 1][bc][bs]);
      async_ld_b128(lb, srcB0 + kn);
      async_ld_b128_o16(lb, srcB0 + kn);
      if (kn + 32 < K) __builtin_prefetch(srcA0 + kn + 32, 0, 1);
    }

    // ---- fragments from current buffer ----
    HFrag af[2], bf[4];
#pragma unroll
    for (int i = 0; i < 2; ++i) {
      const int row = wm * 32 + i * 16 + lam;
      af[i].h[0] = *(const v8h*)&As[buf][row][8 * hi];
      af[i].h[1] = *(const v8h*)&As[buf][row][16 + 8 * hi];
    }
#pragma unroll
    for (int j = 0; j < 4; ++j) {
      const int col = wn * 64 + j * 16 + lam;
      bf[j].h[0] = *(const v8h*)&Bt[buf][col][8 * hi];
      bf[j].h[1] = *(const v8h*)&Bt[buf][col][16 + 8 * hi];
    }
#pragma unroll
    for (int i = 0; i < 2; ++i)
#pragma unroll
      for (int j = 0; j < 4; ++j)
        acc[i][j] = wmma16(af[i].v, bf[j].v, acc[i][j]);
  }

  // --- epilogue ---
#pragma unroll
  for (int i = 0; i < 2; ++i) {
#pragma unroll
    for (int j = 0; j < 4; ++j) {
      const int cbase = n0 + wn * 64 + j * 16 + lam;
      const float bv = bias ? bias[cbase] : 0.0f;
#pragma unroll
      for (int r = 0; r < 8; ++r) {
        const int row = m0 + wm * 32 + i * 16 + r + 8 * hi;
        float v = acc[i][j][r] + bv;
        if (doGelu) v = gelu_exact(v);
        const size_t idx = (size_t)row * N + cbase;
        if (residual) v += residual[idx];
        if (outF32) outF32[idx] = v;
        if (outF16) outF16[idx] = (_Float16)v;
        if (outF16T) {
          // [row/1024][N][1024]  (per-batch transposed, used for V)
          const size_t tix = ((size_t)(row >> 10) * N + cbase) * SS + (row & 1023);
          outF16T[tix] = (_Float16)v;
        }
      }
    }
  }
}

// ---------------------------------------------------------------------------
// Flash-style attention.  Grid: B*H*(S/128) blocks of 256 threads (8 waves).
// Each wave owns 16 query rows.  Two passes over the 1024 keys in tiles of 32:
//  pass 1: streaming row max / sum of exp (K only)
//  pass 2: P = exp(s*scale - m)/l  ->  ctx += P @ V   (WMMA via LDS re-layout)
// K and V tiles staged via async-to-LDS, double buffered. V comes pre-
// transposed from the V-projection GEMM (VtG layout [b][col][s]).
// ---------------------------------------------------------------------------
__global__ void attention_wmma(const _Float16* __restrict__ Q,
                               const _Float16* __restrict__ Kv,
                               const _Float16* __restrict__ VtG,
                               float* __restrict__ ctx) {
  const int bid  = blockIdx.x;
  const int qblk = bid & 7;            // 8 q-blocks of 128 rows
  const int h    = (bid >> 3) % HH;
  const int b    = bid / (8 * HH);
  const int t    = threadIdx.x;
  const int wave = t >> 5;
  const int lane = t & 31;
  const int lam  = lane & 15;
  const int hi   = lane >> 4;

  const size_t base = ((size_t)b * SS) * DD + (size_t)h * HDD; // K/Q head base
  const _Float16* vbase = VtG + ((size_t)b * DD + h * HDD) * SS; // [64][1024]
  const int q0 = qblk * 128 + wave * 16;                       // wave's q rows
  const float scale = 0.125f;                                  // 64^-0.5

  __shared__ __align__(16) _Float16 Ks[2][32][64];    // 8 KB  [key][d]
  __shared__ __align__(16) _Float16 Vt[2][64][32];    // 8 KB  [d][key]
  __shared__ __align__(16) _Float16 Pb[8][16][32];    // 8 KB  per-wave P

  // staging coordinates
  const int ldk = t >> 3;          // 0..31  key row (K tile)
  const int ldg = (t & 7) * 8;     // d-offset (8 halves)
  const int vd  = t >> 2;          // 0..63  d row (V tile)
  const int vs  = (t & 3) * 8;     // key-offset (8 halves)

  auto stageK = [&](int bufi, int kt) {
    async_ld_b128(lds_off(&Ks[bufi][ldk][ldg]),
                  Kv + base + (size_t)(kt + ldk) * DD + ldg);
  };
  auto stageV = [&](int bufi, int kt) {
    async_ld_b128(lds_off(&Vt[bufi][vd][vs]),
                  vbase + (size_t)vd * SS + kt + vs);
  };

  // Q fragments (A operand, 16 rows x 64 contraction = 2 frags) -- persistent
  HFrag qf[2];
  {
    const _Float16* qrow = Q + base + (size_t)(q0 + lam) * DD;
#pragma unroll
    for (int c = 0; c < 2; ++c) {
      qf[c].h[0] = *(const v8h*)(qrow + c * 32 + 8 * hi);
      qf[c].h[1] = *(const v8h*)(qrow + c * 32 + 16 + 8 * hi);
    }
  }

  float mrow[8], lrow[8];
#pragma unroll
  for (int r = 0; r < 8; ++r) { mrow[r] = -1e30f; lrow[r] = 0.0f; }

  // ---------------- pass 1: softmax statistics (K only) ----------------
  stageK(0, 0);
  int buf = 0;
  for (int kt = 0; kt < SS; kt += 32, buf ^= 1) {
    wait_async0();
    __syncthreads();
    if (kt + 32 < SS) stageK(buf ^ 1, kt + 32);
#pragma unroll
    for (int kt2 = 0; kt2 < 2; ++kt2) {
      HFrag bf0, bf1;
      const int key = kt2 * 16 + lam;
      bf0.h[0] = *(const v8h*)&Ks[buf][key][8 * hi];
      bf0.h[1] = *(const v8h*)&Ks[buf][key][16 + 8 * hi];
      bf1.h[0] = *(const v8h*)&Ks[buf][key][32 + 8 * hi];
      bf1.h[1] = *(const v8h*)&Ks[buf][key][32 + 16 + 8 * hi];
      v8f s = {};
      s = wmma16(qf[0].v, bf0.v, s);
      s = wmma16(qf[1].v, bf1.v, s);
#pragma unroll
      for (int r = 0; r < 8; ++r) {
        float v = s[r] * scale;
        float mx = v;
        mx = fmaxf(mx, __shfl_xor(mx, 8, 16));
        mx = fmaxf(mx, __shfl_xor(mx, 4, 16));
        mx = fmaxf(mx, __shfl_xor(mx, 2, 16));
        mx = fmaxf(mx, __shfl_xor(mx, 1, 16));
        const float mnew = fmaxf(mrow[r], mx);
        float e = __expf(v - mnew);
        e += __shfl_xor(e, 8, 16);
        e += __shfl_xor(e, 4, 16);
        e += __shfl_xor(e, 2, 16);
        e += __shfl_xor(e, 1, 16);
        lrow[r] = lrow[r] * __expf(mrow[r] - mnew) + e;
        mrow[r] = mnew;
      }
    }
  }

  float inv[8];
#pragma unroll
  for (int r = 0; r < 8; ++r) inv[r] = 1.0f / lrow[r];

  v8f zero = {};
  v8f cacc[4];
#pragma unroll
  for (int j = 0; j < 4; ++j) cacc[j] = zero;

  // ---------------- pass 2: P @ V ----------------
  __syncthreads();          // pass-1 reads done before restaging buffer 0
  stageK(0, 0);
  stageV(0, 0);
  buf = 0;
  for (int kt = 0; kt < SS; kt += 32, buf ^= 1) {
    wait_async0();
    __syncthreads();
    if (kt + 32 < SS) { stageK(buf ^ 1, kt + 32); stageV(buf ^ 1, kt + 32); }
#pragma unroll
    for (int kt2 = 0; kt2 < 2; ++kt2) {
      HFrag bf0, bf1;
      const int key = kt2 * 16 + lam;
      bf0.h[0] = *(const v8h*)&Ks[buf][key][8 * hi];
      bf0.h[1] = *(const v8h*)&Ks[buf][key][16 + 8 * hi];
      bf1.h[0] = *(const v8h*)&Ks[buf][key][32 + 8 * hi];
      bf1.h[1] = *(const v8h*)&Ks[buf][key][32 + 16 + 8 * hi];
      v8f s = {};
      s = wmma16(qf[0].v, bf0.v, s);
      s = wmma16(qf[1].v, bf1.v, s);
#pragma unroll
      for (int r = 0; r < 8; ++r) {
        const float p = __expf(s[r] * scale - mrow[r]) * inv[r];
        Pb[wave][r + 8 * hi][kt2 * 16 + lam] = (_Float16)p;
      }
    }
    __syncthreads();   // P visible (also orders LDS write->read within wave)

    HFrag pf;
    pf.h[0] = *(const v8h*)&Pb[wave][lam][8 * hi];
    pf.h[1] = *(const v8h*)&Pb[wave][lam][16 + 8 * hi];
#pragma unroll
    for (int j = 0; j < 4; ++j) {
      HFrag vf;
      const int dcol = j * 16 + lam;
      vf.h[0] = *(const v8h*)&Vt[buf][dcol][8 * hi];
      vf.h[1] = *(const v8h*)&Vt[buf][dcol][16 + 8 * hi];
      cacc[j] = wmma16(pf.v, vf.v, cacc[j]);
    }
  }

  // store ctx (f32, [B*S, D] with head offset)
#pragma unroll
  for (int j = 0; j < 4; ++j) {
#pragma unroll
    for (int r = 0; r < 8; ++r) {
      const int row = q0 + r + 8 * hi;
      const int col = h * HDD + j * 16 + lam;
      ctx[((size_t)b * SS + row) * DD + col] = cacc[j][r];
    }
  }
}

// ---------------------------------------------------------------------------
// LayerNorm over rows of 768.  out = [residual +] LN(in)*g + b
// one block (256 thr) per row; optional f32 and/or f16 outputs.
// ---------------------------------------------------------------------------
__global__ void layernorm_row(const float* __restrict__ in,
                              const float* __restrict__ res,
                              const float* __restrict__ g,
                              const float* __restrict__ bta,
                              float* __restrict__ outF32,
                              _Float16* __restrict__ outF16) {
  const int row = blockIdx.x;
  const int t = threadIdx.x;
  const float* x = in + (size_t)row * DD;
  __shared__ float red[256];

  float s = 0.0f;
#pragma unroll
  for (int i = 0; i < 3; ++i) s += x[t + i * 256];
  red[t] = s;
  __syncthreads();
  for (int o = 128; o > 0; o >>= 1) {
    if (t < o) red[t] += red[t + o];
    __syncthreads();
  }
  const float mu = red[0] * (1.0f / DD);
  __syncthreads();

  s = 0.0f;
#pragma unroll
  for (int i = 0; i < 3; ++i) {
    const float d = x[t + i * 256] - mu;
    s += d * d;
  }
  red[t] = s;
  __syncthreads();
  for (int o = 128; o > 0; o >>= 1) {
    if (t < o) red[t] += red[t + o];
    __syncthreads();
  }
  const float rstd = rsqrtf(red[0] * (1.0f / DD) + 1e-5f);

#pragma unroll
  for (int i = 0; i < 3; ++i) {
    const int c = t + i * 256;
    float v = (x[c] - mu) * rstd * g[c] + bta[c];
    const size_t idx = (size_t)row * DD + c;
    if (res) v += res[idx];
    if (outF32) outF32[idx] = v;
    if (outF16) outF16[idx] = (_Float16)v;
  }
}

// ---------------------------------------------------------------------------
// Host-side orchestration
// ---------------------------------------------------------------------------
extern "C" void kernel_launch(void* const* d_in, const int* in_sizes, int n_in,
                              void* d_out, int out_size, void* d_ws, size_t ws_size,
                              hipStream_t stream) {
  const float* x     = (const float*)d_in[0];
  const float* wq    = (const float*)d_in[1];
  const float* bq    = (const float*)d_in[2];
  const float* wk    = (const float*)d_in[3];
  const float* bk    = (const float*)d_in[4];
  const float* wv    = (const float*)d_in[5];
  const float* bv    = (const float*)d_in[6];
  const float* ln1_g = (const float*)d_in[7];
  const float* ln1_b = (const float*)d_in[8];
  const float* ln2_g = (const float*)d_in[9];
  const float* ln2_b = (const float*)d_in[10];
  const float* w_in  = (const float*)d_in[11];
  const float* b_in  = (const float*)d_in[12];
  const float* w_out = (const float*)d_in[13];
  const float* b_out = (const float*)d_in[14];
  float* out = (float*)d_out;

  // ---- workspace carve ----
  size_t off = 0;
  auto carve = [&](size_t bytes) -> void* {
    void* p = (char*)d_ws + off;
    off += (bytes + 255) & ~(size_t)255;
    return p;
  };
  _Float16* xf16    = (_Float16*)carve((size_t)ROWS * DD * 2);
  _Float16* wq16t   = (_Float16*)carve((size_t)DD * DD * 2);   // [N][K]
  _Float16* wk16t   = (_Float16*)carve((size_t)DD * DD * 2);
  _Float16* wv16t   = (_Float16*)carve((size_t)DD * DD * 2);
  _Float16* win16t  = (_Float16*)carve((size_t)DD * HID * 2);  // [3072][768]
  _Float16* wout16t = (_Float16*)carve((size_t)HID * DD * 2);  // [768][3072]
  _Float16* Q16     = (_Float16*)carve((size_t)ROWS * DD * 2);
  _Float16* K16     = (_Float16*)carve((size_t)ROWS * DD * 2);
  _Float16* VtG     = (_Float16*)carve((size_t)ROWS * DD * 2); // [b][col][s]
  float*    ctx     = (float*)   carve((size_t)ROWS * DD * 4);
  float*    xres    = (float*)   carve((size_t)ROWS * DD * 4);
  _Float16* y16     = (_Float16*)carve((size_t)ROWS * DD * 2);
  _Float16* h1      = (_Float16*)carve((size_t)ROWS * HID * 2);
  (void)ws_size; (void)n_in; (void)in_sizes; (void)out_size;

  // ---- 1. casts (x plain; weights cast + transpose to [N][K]) ----
  cast_f32_to_f16<<<1024, 256, 0, stream>>>(x, xf16, (long)ROWS * DD);
  {
    dim3 gT(DD / 32, DD / 32);
    cast_transpose_f32_to_f16<<<gT, 256, 0, stream>>>(wq, wq16t, DD, DD);
    cast_transpose_f32_to_f16<<<gT, 256, 0, stream>>>(wk, wk16t, DD, DD);
    cast_transpose_f32_to_f16<<<gT, 256, 0, stream>>>(wv, wv16t, DD, DD);
    dim3 gTi(HID / 32, DD / 32);   // w_in  [768,3072] -> [3072][768]
    cast_transpose_f32_to_f16<<<gTi, 256, 0, stream>>>(w_in, win16t, DD, HID);
    dim3 gTo(DD / 32, HID / 32);   // w_out [3072,768] -> [768][3072]
    cast_transpose_f32_to_f16<<<gTo, 256, 0, stream>>>(w_out, wout16t, HID, DD);
  }

  // ---- 2. QKV projections (WMMA, async staged) ----
  dim3 gQKV(DD / 128, ROWS / 128);
  gemm_wmma_f16<<<gQKV, 256, 0, stream>>>(xf16, wq16t, bq, nullptr, Q16, nullptr,
                                          nullptr, ROWS, DD, DD, 0);
  gemm_wmma_f16<<<gQKV, 256, 0, stream>>>(xf16, wk16t, bk, nullptr, K16, nullptr,
                                          nullptr, ROWS, DD, DD, 0);
  gemm_wmma_f16<<<gQKV, 256, 0, stream>>>(xf16, wv16t, bv, nullptr, nullptr, VtG,
                                          nullptr, ROWS, DD, DD, 0);

  // ---- 3. attention (WMMA flash-style, double-buffered async staging) ----
  attention_wmma<<<BB * HH * (SS / 128), 256, 0, stream>>>(Q16, K16, VtG, ctx);

  // ---- 4. x_res = x + LN1(ctx) ----
  layernorm_row<<<ROWS, 256, 0, stream>>>(ctx, x, ln1_g, ln1_b, xres, nullptr);

  // ---- 5. y = LN2(x_res) (f16 for next GEMM) ----
  layernorm_row<<<ROWS, 256, 0, stream>>>(xres, nullptr, ln2_g, ln2_b, nullptr, y16);

  // ---- 6. h1 = gelu(y @ w_in + b_in) ----
  dim3 gMLP1(HID / 128, ROWS / 128);
  gemm_wmma_f16<<<gMLP1, 256, 0, stream>>>(y16, win16t, b_in, nullptr, h1, nullptr,
                                           nullptr, ROWS, HID, DD, 1);

  // ---- 7. out = x_res + gelu(h1 @ w_out + b_out) ----
  dim3 gMLP2(DD / 128, ROWS / 128);
  gemm_wmma_f16<<<gMLP2, 256, 0, stream>>>(h1, wout16t, b_out, out, nullptr,
                                           nullptr, xres, ROWS, DD, HID, 1);
}
